// GraphRecMultiCo_41308995453069
// MI455X (gfx1250) — matrix operand order repaired and checked by reading, not compiled, not run on hardware
//
#include <hip/hip_runtime.h>
#include <hip/hip_bf16.h>

// ---------------------------------------------------------------------------
// GraphRecMultiCo for MI455X (gfx1250, wave32, WMMA + async-to-LDS).
// N = 2*B*L = 65536 fused rows (src stream then dst stream), D = 128.
//   1) cooc histogram -> app counts            (LDS histograms, 1000 bins)
//   2) Mw = w2 @ proj_w, bb = 2*b2@proj_w+proj_b
//   3) pack weights into WMMA B-fragment layout (f16)
//   4) E = relu(a0*w1+b1)+relu(a1*w1+b1)
//   5) GEMM1: x = E @ Mw + bb  -> d_out (f32)
//   6) LN1 -> H (f16)
//   7) GEMM2: QKV = H @ w_qkv + b_qkv (f16)
//   8) flash attention (async K/Q staging to LDS, WMMA S=QK^T and O=PV)
//   9) GEMM3: x += CTX @ w_o + b_o
//  10) LN2 -> H
//  11) GEMM4: G = gelu(H @ w_ff1 + b_ff1) (f16)
//  12) GEMM5: x += G @ w_ff2 + b_ff2
// GEMM waves are blocked 16x64 (4 WMMA tiles per A fragment load).
// ---------------------------------------------------------------------------

typedef __attribute__((ext_vector_type(16))) _Float16 v16h;
typedef __attribute__((ext_vector_type(8)))  float    v8f;

union frag16 { uint4 u[2]; v16h v; };

__device__ __forceinline__ v16h load_frag16(const _Float16* p0, const _Float16* p1) {
    frag16 f;
    f.u[0] = *(const uint4*)p0;
    f.u[1] = *(const uint4*)p1;
    return f.v;
}

__device__ __forceinline__ unsigned packh2(float lo, float hi) {
    union { _Float16 h[2]; unsigned u; } p;
    p.h[0] = (_Float16)lo; p.h[1] = (_Float16)hi;
    return p.u;
}

// CDNA5 async global->LDS copy (16B per lane), tracked by ASYNCcnt.
__device__ __forceinline__ void async_copy_b128(void* lds_ptr, const void* gptr) {
    unsigned loff = (unsigned)(uintptr_t)lds_ptr;   // low 32 bits = LDS byte offset
    asm volatile("global_load_async_to_lds_b128 %0, %1, off"
                 :: "v"(loff), "v"(gptr)
                 : "memory");
}
__device__ __forceinline__ void wait_asynccnt0() {
    asm volatile("s_wait_asynccnt 0x0" ::: "memory");
}

#define NROWS 65536   // 2*B*L
#define DMODEL 128
#define LSEQ 512

// ---------------------------------------------------------------------------
// 1) co-occurrence counts via per-row LDS histograms
// ---------------------------------------------------------------------------
__global__ void cooc_kernel(const int* __restrict__ src, const int* __restrict__ dst,
                            float2* __restrict__ app) {
    __shared__ int hs[1000];
    __shared__ int hd[1000];
    const int b = blockIdx.x;
    const int t = threadIdx.x;
    for (int i = t; i < 1000; i += 256) { hs[i] = 0; hd[i] = 0; }
    __syncthreads();
    for (int j = t; j < LSEQ; j += 256) {
        atomicAdd(&hs[src[b * LSEQ + j]], 1);
        atomicAdd(&hd[dst[b * LSEQ + j]], 1);
    }
    __syncthreads();
    for (int j = t; j < LSEQ; j += 256) {
        int si = src[b * LSEQ + j];
        int di = dst[b * LSEQ + j];
        float2 as, ad;
        as.x = (si == 0) ? 0.f : (float)hs[si];
        as.y = (si == 0) ? 0.f : (float)hd[si];
        ad.x = (di == 0) ? 0.f : (float)hs[di];
        ad.y = (di == 0) ? 0.f : (float)hd[di];
        app[b * LSEQ + j] = as;
        app[32768 + b * LSEQ + j] = ad;
    }
}

// ---------------------------------------------------------------------------
// 2) Mw = w2 @ proj_w (128x128), bb = 2*b2 @ proj_w + proj_b
// ---------------------------------------------------------------------------
__global__ void mw_kernel(const float* __restrict__ w2, const float* __restrict__ proj_w,
                          const float* __restrict__ b2, const float* __restrict__ proj_b,
                          float* __restrict__ Mw, float* __restrict__ bb) {
    int d = threadIdx.x;
    int i = blockIdx.x;
    if (i < 128) {
        float s = 0.f;
        for (int j = 0; j < 128; ++j) s += w2[i * 128 + j] * proj_w[j * 128 + d];
        Mw[i * 128 + d] = s;
    } else {
        float s = proj_b[d];
        for (int j = 0; j < 128; ++j) s += 2.0f * b2[j] * proj_w[j * 128 + d];
        bb[d] = s;
    }
}

// ---------------------------------------------------------------------------
// 3) pack f32 weight (K x N row-major) into WMMA B-fragment layout (f16)
// ---------------------------------------------------------------------------
__global__ void pack_b_kernel(const float* __restrict__ W, unsigned* __restrict__ P,
                              int K, int N) {
    int idx = blockIdx.x * blockDim.x + threadIdx.x;
    if (idx >= (K * N) / 2) return;
    int i    = idx & 7;
    int lane = (idx >> 3) & 31;
    int rest = idx >> 8;
    int KT = K >> 5;
    int kt = rest % KT;
    int ct = rest / KT;
    int col = ct * 16 + (lane & 15);
    int k0  = kt * 32 + ((i >> 2) << 4) + ((lane >> 4) << 3) + 2 * (i & 3);
    P[idx] = packh2(W[(size_t)k0 * N + col], W[(size_t)(k0 + 1) * N + col]);
}

// ---------------------------------------------------------------------------
// 4) E = relu(a0*w1 + b1) + relu(a1*w1 + b1), f16 out (NROWS x 128)
// ---------------------------------------------------------------------------
__global__ void encode_kernel(const float2* __restrict__ app, const float* __restrict__ w1,
                              const float* __restrict__ b1, _Float16* __restrict__ E) {
    size_t idx = (size_t)blockIdx.x * blockDim.x + threadIdx.x;
    int d = (int)(idx & 127);
    size_t n = idx >> 7;
    float2 a = app[n];
    float w = w1[d], bi = b1[d];
    float e = fmaxf(fmaf(a.x, w, bi), 0.f) + fmaxf(fmaf(a.y, w, bi), 0.f);
    E[idx] = (_Float16)e;
}

// ---------------------------------------------------------------------------
// Generic WMMA GEMM: C(N x Dout) = A(N x K f16 row-major) @ Wpacked + bias.
// Each wave owns a 16x64 output tile: one A fragment feeds 4 WMMAs.
// OUT_MODE: 0 = store f32, 1 = add into f32 (residual), 2 = store f16
// ---------------------------------------------------------------------------
template <int K, int OUT_MODE, bool GELU>
__global__ void gemm_kernel(const _Float16* __restrict__ A, const unsigned* __restrict__ Bp,
                            const float* __restrict__ bias, float* __restrict__ Cf,
                            _Float16* __restrict__ C16, int Dout, int colGroups) {
    constexpr int KT = K / 32;
    const int wave = threadIdx.x >> 5;
    const int lane = threadIdx.x & 31;
    const int ln16 = lane & 15;
    const int hi   = lane >> 4;
    const int tid  = blockIdx.x * 8 + wave;
    const int rt  = tid / colGroups;
    const int ct0 = (tid % colGroups) * 4;

    v8f acc[4] = {};
    const _Float16* arow = A + (size_t)(rt * 16 + ln16) * K + hi * 8;
#pragma unroll
    for (int kt = 0; kt < KT; ++kt) {
        v16h a = load_frag16(arow + kt * 32, arow + kt * 32 + 16);
#pragma unroll
        for (int i = 0; i < 4; ++i) {
            const uint4* bp =
                (const uint4*)(Bp + ((size_t)(ct0 + i) * KT + kt) * 256 + lane * 8);
            frag16 f; f.u[0] = bp[0]; f.u[1] = bp[1];
            acc[i] = __builtin_amdgcn_wmma_f32_16x16x32_f16(false, a, false, f.v,
                                                            (short)0, acc[i], false, false);
        }
    }
    const int rowbase = rt * 16 + hi * 8;
#pragma unroll
    for (int i = 0; i < 4; ++i) {
        const int col = (ct0 + i) * 16 + ln16;
        const float bv = bias ? bias[col] : 0.f;
#pragma unroll
        for (int r = 0; r < 8; ++r) {
            float v = acc[i][r] + bv;
            if (GELU) v = 0.5f * v * (1.0f + erff(v * 0.70710678f));
            size_t o = (size_t)(rowbase + r) * Dout + col;
            if (OUT_MODE == 0)      Cf[o] = v;
            else if (OUT_MODE == 1) Cf[o] += v;
            else                    C16[o] = (_Float16)v;
        }
    }
}

// ---------------------------------------------------------------------------
// LayerNorm: f32 in (row of 128) -> f16 out. One wave per row.
// ---------------------------------------------------------------------------
__global__ void ln_kernel(const float* __restrict__ X, const float* __restrict__ g,
                          const float* __restrict__ b, _Float16* __restrict__ Y) {
    const int wave = threadIdx.x >> 5;
    const int lane = threadIdx.x & 31;
    const size_t row = (size_t)blockIdx.x * 8 + wave;
    float4 x = *(const float4*)(X + row * 128 + lane * 4);
    float s = x.x + x.y + x.z + x.w;
    float q = x.x * x.x + x.y * x.y + x.z * x.z + x.w * x.w;
#pragma unroll
    for (int m = 16; m; m >>= 1) {
        s += __shfl_xor(s, m, 32);
        q += __shfl_xor(q, m, 32);
    }
    float mu = s * (1.f / 128.f);
    float var = q * (1.f / 128.f) - mu * mu;
    float rs = rsqrtf(var + 1e-5f);
    int c = lane * 4;
    float xv[4] = {x.x, x.y, x.z, x.w};
#pragma unroll
    for (int i = 0; i < 4; ++i)
        Y[row * 128 + c + i] = (_Float16)((xv[i] - mu) * rs * g[c + i] + b[c + i]);
}

// ---------------------------------------------------------------------------
// Flash attention: one block = (stream*b, head, 16-row q tile), 4 waves.
// Q and K chunks staged to LDS via CDNA5 async copies (ASYNCcnt); V staged
// transposed via VGPR scatter. WMMA for both S = Q K^T and O += P V.
// ---------------------------------------------------------------------------
__global__ void attn_kernel(const _Float16* __restrict__ QKV, _Float16* __restrict__ CTX) {
    const int bx = blockIdx.x;
    const int qt = bx & 31;
    const int h  = (bx >> 5) & 1;
    const int bb = bx >> 6;              // 0..127 (stream*64 + batch)
    const size_t rowb = (size_t)bb * LSEQ;
    const int q0 = qt * 16;

    __shared__ __align__(16) _Float16 Qs[16][64];
    __shared__ __align__(16) _Float16 Ks[64][64];
    __shared__ __align__(16) _Float16 Vts[64][64];   // Vts[dh][key]
    __shared__ float Ss[16][64];
    __shared__ __align__(16) _Float16 Ps[16][64];
    __shared__ float mS[16], lS[16], cS[16];

    const int t = threadIdx.x;           // 128 threads, 4 waves
    const int wave = t >> 5;
    const int lane = t & 31;
    const int ln16 = lane & 15;
    const int hi   = lane >> 4;

    {   // async-stage Q tile (16 x 64); completion covered by first chunk's wait
        int r = t >> 3, seg = t & 7;
        async_copy_b128(&Qs[r][seg * 8],
                        QKV + (rowb + q0 + r) * 384 + h * 64 + seg * 8);
    }
    if (t < 16) { mS[t] = -1e30f; lS[t] = 0.f; }

    v8f o = {};
    for (int kc = 0; kc < 8; ++kc) {
        __syncthreads();   // previous iteration's LDS consumers are done
        const int keyb = kc * 64;
        // stage K chunk (async, 64x64) and V chunk transposed (dh-major)
        for (int it = 0; it < 4; ++it) {
            int sft = t + it * 128;
            int r = sft >> 3, seg = sft & 7;
            async_copy_b128(&Ks[r][seg * 8],
                            QKV + (rowb + keyb + r) * 384 + 128 + h * 64 + seg * 8);
            uint4 vv = *(const uint4*)(QKV + (rowb + keyb + r) * 384 + 256 + h * 64 + seg * 8);
            const _Float16* ve = (const _Float16*)&vv;
#pragma unroll
            for (int e = 0; e < 8; ++e) Vts[seg * 8 + e][r] = ve[e];
        }
        wait_asynccnt0();  // own async copies landed; barrier publishes all waves'
        __syncthreads();
        // S = Q K^T for this wave's 16 keys (scale 1/sqrt(64))
        v8f s = {};
#pragma unroll
        for (int ks = 0; ks < 64; ks += 32) {
            v16h a = load_frag16(&Qs[ln16][ks + hi * 8], &Qs[ln16][ks + 16 + hi * 8]);
            v16h bf = load_frag16(&Ks[wave * 16 + ln16][ks + hi * 8],
                                  &Ks[wave * 16 + ln16][ks + 16 + hi * 8]);
            s = __builtin_amdgcn_wmma_f32_16x16x32_f16(false, a, false, bf,
                                                       (short)0, s, false, false);
        }
#pragma unroll
        for (int r = 0; r < 8; ++r) Ss[hi * 8 + r][wave * 16 + ln16] = s[r] * 0.125f;
        __syncthreads();
        // online softmax over this 64-key chunk
        if (t < 16) {
            float m0 = mS[t];
            float cm = m0;
            for (int j = 0; j < 64; ++j) cm = fmaxf(cm, Ss[t][j]);
            float c = __expf(m0 - cm);
            float sum = 0.f;
            for (int j = 0; j < 64; ++j) {
                float p = __expf(Ss[t][j] - cm);
                Ps[t][j] = (_Float16)p;
                sum += p;
            }
            lS[t] = lS[t] * c + sum;
            mS[t] = cm;
            cS[t] = c;
        }
        __syncthreads();
#pragma unroll
        for (int r = 0; r < 8; ++r) o[r] *= cS[hi * 8 + r];
        // O += P @ V_chunk (this wave's 16 dh columns)
#pragma unroll
        for (int ks = 0; ks < 64; ks += 32) {
            v16h a = load_frag16(&Ps[ln16][ks + hi * 8], &Ps[ln16][ks + 16 + hi * 8]);
            v16h bf = load_frag16(&Vts[wave * 16 + ln16][ks + hi * 8],
                                  &Vts[wave * 16 + ln16][ks + 16 + hi * 8]);
            o = __builtin_amdgcn_wmma_f32_16x16x32_f16(false, a, false, bf,
                                                       (short)0, o, false, false);
        }
    }
#pragma unroll
    for (int r = 0; r < 8; ++r) {
        int rl = hi * 8 + r;
        float inv = 1.0f / lS[rl];
        CTX[(rowb + q0 + rl) * DMODEL + h * 64 + wave * 16 + ln16] = (_Float16)(o[r] * inv);
    }
}

// ---------------------------------------------------------------------------
extern "C" void kernel_launch(void* const* d_in, const int* in_sizes, int n_in,
                              void* d_out, int out_size, void* d_ws, size_t ws_size,
                              hipStream_t stream) {
    const int*   src_ids = (const int*)d_in[0];
    const int*   dst_ids = (const int*)d_in[1];
    const float* w1      = (const float*)d_in[2];
    const float* b1      = (const float*)d_in[3];
    const float* w2      = (const float*)d_in[4];
    const float* b2      = (const float*)d_in[5];
    const float* proj_w  = (const float*)d_in[6];
    const float* proj_b  = (const float*)d_in[7];
    const float* ln1_g   = (const float*)d_in[8];
    const float* ln1_b   = (const float*)d_in[9];
    const float* w_qkv   = (const float*)d_in[10];
    const float* b_qkv   = (const float*)d_in[11];
    const float* w_o     = (const float*)d_in[12];
    const float* b_o     = (const float*)d_in[13];
    const float* ln2_g   = (const float*)d_in[14];
    const float* ln2_b   = (const float*)d_in[15];
    const float* w_ff1   = (const float*)d_in[16];
    const float* b_ff1   = (const float*)d_in[17];
    const float* w_ff2   = (const float*)d_in[18];
    const float* b_ff2   = (const float*)d_in[19];
    float* x = (float*)d_out;    // (2,B,L,D) f32 residual stream, written in full

    // workspace arena (~101 MB)
    char* base = (char*)d_ws;
    size_t off = 0;
    auto alloc = [&](size_t bytes) {
        void* p = base + off;
        off = (off + bytes + 255) & ~(size_t)255;
        return p;
    };
    float2*    app   = (float2*)   alloc((size_t)NROWS * sizeof(float2));      // 512 KB
    _Float16*  E     = (_Float16*) alloc((size_t)NROWS * DMODEL * 2);          // 16 MB (reused as CTX)
    _Float16*  H     = (_Float16*) alloc((size_t)NROWS * DMODEL * 2);          // 16 MB
    _Float16*  QKV   = (_Float16*) alloc((size_t)NROWS * 512 * 2);             // 64 MB (reused as FF1 out)
    float*     Mw    = (float*)    alloc(128 * 128 * 4);
    float*     bbv   = (float*)    alloc(128 * 4);
    unsigned*  Mw_p  = (unsigned*) alloc(128 * 128 / 2 * 4);
    unsigned*  qkv_p = (unsigned*) alloc(128 * 384 / 2 * 4);
    unsigned*  wo_p  = (unsigned*) alloc(128 * 128 / 2 * 4);
    unsigned*  ff1_p = (unsigned*) alloc(128 * 512 / 2 * 4);
    unsigned*  ff2_p = (unsigned*) alloc(512 * 128 / 2 * 4);
    _Float16*  CTX = E;     // E dead after GEMM1
    _Float16*  G   = QKV;   // QKV dead after attention

    // 1) co-occurrence counts
    cooc_kernel<<<64, 256, 0, stream>>>(src_ids, dst_ids, app);
    // 2) fold encoder tail + projection into one matrix
    mw_kernel<<<129, 128, 0, stream>>>(w2, proj_w, b2, proj_b, Mw, bbv);
    // 3) prepack weights into B-fragment layout
    pack_b_kernel<<<32, 256, 0, stream>>>(Mw, Mw_p, 128, 128);
    pack_b_kernel<<<96, 256, 0, stream>>>(w_qkv, qkv_p, 128, 384);
    pack_b_kernel<<<32, 256, 0, stream>>>(w_o, wo_p, 128, 128);
    pack_b_kernel<<<128, 256, 0, stream>>>(w_ff1, ff1_p, 128, 512);
    pack_b_kernel<<<128, 256, 0, stream>>>(w_ff2, ff2_p, 512, 128);
    // 4) scalar-MLP front half
    encode_kernel<<<(NROWS * DMODEL) / 256, 256, 0, stream>>>(app, w1, b1, E);
    // 5) x = E @ Mw + bb   (initial residual, f32 into d_out)
    gemm_kernel<128, 0, false><<<1024, 256, 0, stream>>>(E, Mw_p, bbv, x, nullptr, 128, 2);
    // 6) LN1
    ln_kernel<<<NROWS / 8, 256, 0, stream>>>(x, ln1_g, ln1_b, H);
    // 7) QKV
    gemm_kernel<128, 2, false><<<3072, 256, 0, stream>>>(H, qkv_p, b_qkv, nullptr, QKV, 384, 6);
    // 8) attention
    attn_kernel<<<8192, 128, 0, stream>>>(QKV, CTX);
    // 9) x += CTX @ w_o + b_o
    gemm_kernel<128, 1, false><<<1024, 256, 0, stream>>>(CTX, wo_p, b_o, x, nullptr, 128, 2);
    // 10) LN2
    ln_kernel<<<NROWS / 8, 256, 0, stream>>>(x, ln2_g, ln2_b, H);
    // 11) G = gelu(H @ w_ff1 + b_ff1)
    gemm_kernel<128, 2, true><<<4096, 256, 0, stream>>>(H, ff1_p, b_ff1, nullptr, G, 512, 8);
    // 12) x += G @ w_ff2 + b_ff2
    gemm_kernel<512, 1, false><<<1024, 256, 0, stream>>>(G, ff2_p, b_ff2, x, nullptr, 128, 2);
}
